// NeRFRenderer_85083302134030
// MI455X (gfx1250) — matrix operand order repaired and checked by reading, not compiled, not run on hardware
//
#include <hip/hip_runtime.h>
#include <hip/hip_bf16.h>

typedef __attribute__((ext_vector_type(16))) _Float16 v16h;
typedef __attribute__((ext_vector_type(2)))  _Float16 v2h;
typedef __attribute__((ext_vector_type(8)))  float    v8f;

#define NRAYS   32768
#define KS      64
#define NPT     (NRAYS * KS)      // 2,097,152 points
#define HDIM    128
#define STR     136               // padded LDS row stride (halves) -> conflict-free
#define NBLK    1024
#define RAYS_PER_BLK (NRAYS / NBLK)   // 32
#define ITERS   (RAYS_PER_BLK / 2)    // 16 (2 rays / iteration)
#define OBSTR   (KS * 4 + 5)          // 261 floats: outbuf ray stride, coprime with 64 banks

// ---------------- monotonic float<->uint key for ordered atomics ----------------
__device__ __forceinline__ unsigned fkey(float f) {
    unsigned u = __float_as_uint(f);
    return (u & 0x80000000u) ? ~u : (u | 0x80000000u);
}
__device__ __forceinline__ float keyf(unsigned k) {
    unsigned u = (k & 0x80000000u) ? (k & 0x7fffffffu) : ~k;
    return __uint_as_float(u);
}

// ---------------- WMMA fragment loaders (wave32, 16x16x32 f16) ----------------
// A (16x32, MxK): lane = g*16+ln, row M=ln; VGPR v half h: K = 16g + 2v + h
__device__ __forceinline__ v16h load_Afrag(const _Float16* base, int ln, int g, int kk) {
    const _Float16* row = base + ln * STR + kk * 32 + 16 * g;
    v16h a;
#pragma unroll
    for (int v = 0; v < 8; ++v) {
        int off = 2 * v;
        a[2 * v]     = row[off];
        a[2 * v + 1] = row[off + 1];
    }
    return a;
}
// B (32x16, KxN) from transposed weight LDS wt[n][k]: lane: N row = nrow,
// VGPR v half h: K = kk*32 + 16g + 2v + h  -> contiguous 16 halves in row n.
__device__ __forceinline__ v16h load_Bfrag(const _Float16* wt, int nrow, int g, int kk) {
    const _Float16* row = wt + nrow * STR + kk * 32 + 16 * g;
    v16h b;
#pragma unroll
    for (int i = 0; i < 16; ++i) b[i] = row[i];
    return b;
}

// ---------------- init accumulators ----------------
__global__ void init_stats(float* stats) {
    if (threadIdx.x == 0) {
        stats[0] = 0.f;                               // S = sum(sigma)
        stats[1] = 0.f;                               // Z = sum(exp(s-mx))
        stats[2] = 0.f;                               // loss sum
        ((unsigned*)stats)[3] = 0u;                   // max key (== -inf)
        ((unsigned*)stats)[4] = 0xFFFFFFFFu;          // min key (== +inf)
    }
}

// ---------------- fused MLP (WMMA) + compositing ----------------
__global__ __launch_bounds__(256) void nerf_fused(
    const float* __restrict__ rays, const float* __restrict__ z_samp,
    const float* __restrict__ W1, const float* __restrict__ b1,
    const float* __restrict__ W2, const float* __restrict__ b2,
    const float* __restrict__ W3, const float* __restrict__ b3,
    float* __restrict__ rgb_final, float* __restrict__ depth_final,
    float* __restrict__ weights_out, float* __restrict__ ws_sigma,
    float* __restrict__ stats)
{
    __shared__ _Float16 hbuf[128 * STR];     // h1 then h2 (per-wave row ownership)
    __shared__ _Float16 w2t[HDIM * STR];     // W2 transposed [n][k], f16
    __shared__ _Float16 w3t[16 * STR];       // W3 transposed+padded [n(16)][k], f16
    __shared__ float w1s[3 * HDIM];
    __shared__ float b1s[HDIM], b2s[HDIM], b3s[4];
    __shared__ float outbuf[RAYS_PER_BLK * OBSTR];  // all layer-3 outputs of the block
    __shared__ float red[256];

    const int tid  = threadIdx.x;
    const int wave = tid >> 5;
    const int lane = tid & 31;
    const int ln   = lane & 15;
    const int g    = lane >> 4;
    const int r_base = blockIdx.x * RAYS_PER_BLK;

    // hint the ray/z data into cache early (global_prefetch_b8)
    __builtin_prefetch(rays + (r_base + (tid & 31)) * 8, 0, 1);
    __builtin_prefetch(z_samp + (r_base + (tid & 31)) * KS, 0, 1);

    // ---- preload weights into LDS (f32 -> f16, transposed) ----
    for (int i = tid; i < 3 * HDIM; i += 256) w1s[i] = W1[i];
    for (int i = tid; i < HDIM; i += 256) { b1s[i] = b1[i]; b2s[i] = b2[i]; }
    if (tid < 4) b3s[tid] = b3[tid];
    for (int i = tid; i < HDIM * HDIM; i += 256) {
        int k = i >> 7, n = i & 127;
        w2t[n * STR + k] = (_Float16)W2[i];
    }
    for (int i = tid; i < 16 * STR; i += 256) w3t[i] = (_Float16)0.f;
    __syncthreads();
    for (int i = tid; i < HDIM * 4; i += 256) {
        int k = i >> 2, n = i & 3;
        w3t[n * STR + k] = (_Float16)W3[i];
    }
    __syncthreads();

    // ======== main loop: fully barrier-free (strict per-wave LDS row ownership) ========
    for (int it = 0; it < ITERS; ++it) {
        const int r0 = r_base + it * 2;

        // ---- layer 1: h1 = relu(P @ W1 + b1); lane computes a packed feature pair ----
        {
            const int p   = wave * 16 + ln;       // local point 0..127 (own wave's rows)
            const int ray = r0 + (p >> 6);
            const int k   = p & 63;
            const float* rp = rays + ray * 8;
            const float z = z_samp[ray * KS + k];
            const float x  = rp[0] + z * rp[3];
            const float y  = rp[1] + z * rp[4];
            const float zz = rp[2] + z * rp[5];
            _Float16* hrow = hbuf + p * STR;
#pragma unroll 8
            for (int j = 0; j < 32; ++j) {
                const int f = 2 * g + 4 * j;
                float h0 = b1s[f]     + x * w1s[f]     + y * w1s[HDIM + f]     + zz * w1s[2 * HDIM + f];
                float h1 = b1s[f + 1] + x * w1s[f + 1] + y * w1s[HDIM + f + 1] + zz * w1s[2 * HDIM + f + 1];
                v2h pr;
                pr[0] = (_Float16)fmaxf(h0, 0.f);
                pr[1] = (_Float16)fmaxf(h1, 0.f);
                *(v2h*)(hrow + f) = pr;     // packed ds_store_b32, conflict-free
            }
        }

        // ---- layers 2 & 3: WMMA, one wave = 16 points ----
        {
            const int wbase = wave * 16;
            const _Float16* hb = hbuf + wbase * STR;

            v16h A[4];
#pragma unroll
            for (int kk = 0; kk < 4; ++kk) A[kk] = load_Afrag(hb, ln, g, kk);

            // layer 2: 8 N-tiles x 4 k-steps; overwrite own hbuf rows with h2 (f16)
#pragma unroll
            for (int nt = 0; nt < 8; ++nt) {
                v8f acc = {};
#pragma unroll
                for (int kk = 0; kk < 4; ++kk) {
                    v16h Bf = load_Bfrag(w2t, nt * 16 + ln, g, kk);
                    acc = __builtin_amdgcn_wmma_f32_16x16x32_f16(
                        false, A[kk], false, Bf, (short)0, acc, false, false);
                }
                const int n = nt * 16 + ln;
                const float bias = b2s[n];
#pragma unroll
                for (int r = 0; r < 8; ++r) {
                    float v = fmaxf(acc[r] + bias, 0.f);
                    hbuf[(wbase + r + 8 * g) * STR + n] = (_Float16)v;
                }
            }

            // layer 3: out = h2 @ W3pad + b3 (only n<4 meaningful)
            v16h A3[4];
#pragma unroll
            for (int kk = 0; kk < 4; ++kk) A3[kk] = load_Afrag(hb, ln, g, kk);
            v8f acc = {};
#pragma unroll
            for (int kk = 0; kk < 4; ++kk) {
                v16h Bf = load_Bfrag(w3t, ln, g, kk);
                acc = __builtin_amdgcn_wmma_f32_16x16x32_f16(
                    false, A3[kk], false, Bf, (short)0, acc, false, false);
            }
            if (ln < 4) {
                const float bias = b3s[ln];
#pragma unroll
                for (int r = 0; r < 8; ++r) {
                    const int p  = wbase + r + 8 * g;             // local point
                    const int rl = it * 2 + (p >> 6);             // ray-in-block
                    const int k  = p & 63;
                    outbuf[rl * OBSTR + k * 4 + ln] = acc[r] + bias;
                }
            }
        }
    }
    __syncthreads();   // everything for the block's 32 rays is now in outbuf

    // ---- sigma export + running global reductions (all 256 threads) ----
    float loc_sum = 0.f, loc_min = 3.0e38f, loc_max = -3.0e38f;
    for (int p = tid; p < RAYS_PER_BLK * KS; p += 256) {
        const int rl = p >> 6, k = p & 63;
        const float s = outbuf[rl * OBSTR + k * 4 + 3];
        ws_sigma[r_base * KS + p] = s;       // coalesced
        loc_sum += s;
        loc_min = fminf(loc_min, s);
        loc_max = fmaxf(loc_max, s);
    }

    // ---- compositing: 32 rays in parallel, one thread each ----
    if (tid < RAYS_PER_BLK) {
        const int ray = r_base + tid;
        const float* zr = z_samp + ray * KS;
        float* wout = weights_out + ray * KS;
        const float* ob = outbuf + tid * OBSTR;
        float T = 1.f, rf = 0.f, gf = 0.f, bf = 0.f, df = 0.f;
        float zc = zr[0];
        for (int k = 0; k < KS; ++k) {
            const float zn = (k < KS - 1) ? zr[k + 1] : 0.f;
            const float delta = (k < KS - 1) ? (zn - zc) : 1e10f;
            const float* o = ob + k * 4;
            const float s = o[3];
            const float alpha = 1.f - __expf(-fabsf(delta) * fmaxf(s, 0.f));
            const float w = alpha * T;
            T *= (1.f - alpha + 1e-10f);
            wout[k] = w;
            rf += w * (1.f / (1.f + __expf(-o[0])));
            gf += w * (1.f / (1.f + __expf(-o[1])));
            bf += w * (1.f / (1.f + __expf(-o[2])));
            df += w * zc;
            zc = zn;
        }
        rgb_final[ray * 3 + 0] = rf;
        rgb_final[ray * 3 + 1] = gf;
        rgb_final[ray * 3 + 2] = bf;
        depth_final[ray] = df;
    }
    __syncthreads();

    // ---- block reductions -> global atomics ----
    red[tid] = loc_sum; __syncthreads();
    for (int s = 128; s > 0; s >>= 1) { if (tid < s) red[tid] += red[tid + s]; __syncthreads(); }
    if (tid == 0) atomicAdd(&stats[0], red[0]);
    __syncthreads();
    red[tid] = loc_max; __syncthreads();
    for (int s = 128; s > 0; s >>= 1) { if (tid < s) red[tid] = fmaxf(red[tid], red[tid + s]); __syncthreads(); }
    if (tid == 0) atomicMax(((unsigned*)stats) + 3, fkey(red[0]));
    __syncthreads();
    red[tid] = loc_min; __syncthreads();
    for (int s = 128; s > 0; s >>= 1) { if (tid < s) red[tid] = fminf(red[tid], red[tid + s]); __syncthreads(); }
    if (tid == 0) atomicMin(((unsigned*)stats) + 4, fkey(red[0]));
}

// ---------------- pass 2: Z = sum exp(s - max) ----------------
__global__ __launch_bounds__(256) void z_pass(const float* __restrict__ sig,
                                              float* __restrict__ stats) {
    __shared__ float red[256];
    const float mx = keyf(((const unsigned*)stats)[3]);
    const float4* s4 = (const float4*)sig;
    float acc = 0.f;
    for (int i = blockIdx.x * blockDim.x + threadIdx.x; i < NPT / 4; i += gridDim.x * blockDim.x) {
        const float4 v = s4[i];
        acc += __expf(v.x - mx) + __expf(v.y - mx) + __expf(v.z - mx) + __expf(v.w - mx);
    }
    red[threadIdx.x] = acc; __syncthreads();
    for (int s = 128; s > 0; s >>= 1) { if (threadIdx.x < s) red[threadIdx.x] += red[threadIdx.x + s]; __syncthreads(); }
    if (threadIdx.x == 0) atomicAdd(&stats[1], red[0]);
}

// ---------------- pass 3: loss sum = sum (s - sharp)^2 ----------------
// sharp_i = exp(s_i-mx)/Z * S - m,  m = min_i p_i*S = (S>=0 ? S*exp(mn-mx)/Z : S/Z)
__global__ __launch_bounds__(256) void loss_pass(const float* __restrict__ sig,
                                                 float* __restrict__ stats) {
    __shared__ float red[256];
    const float S  = stats[0];
    const float Z  = stats[1];
    const float mx = keyf(((const unsigned*)stats)[3]);
    const float mn = keyf(((const unsigned*)stats)[4]);
    const float m  = (S >= 0.f) ? S * __expf(mn - mx) / Z : S / Z;
    const float SoZ = S / Z;
    const float4* s4 = (const float4*)sig;
    float acc = 0.f;
    for (int i = blockIdx.x * blockDim.x + threadIdx.x; i < NPT / 4; i += gridDim.x * blockDim.x) {
        const float4 v = s4[i];
        float d0 = v.x - (__expf(v.x - mx) * SoZ - m);
        float d1 = v.y - (__expf(v.y - mx) * SoZ - m);
        float d2 = v.z - (__expf(v.z - mx) * SoZ - m);
        float d3 = v.w - (__expf(v.w - mx) * SoZ - m);
        acc += d0 * d0 + d1 * d1 + d2 * d2 + d3 * d3;
    }
    red[threadIdx.x] = acc; __syncthreads();
    for (int s = 128; s > 0; s >>= 1) { if (threadIdx.x < s) red[threadIdx.x] += red[threadIdx.x + s]; __syncthreads(); }
    if (threadIdx.x == 0) atomicAdd(&stats[2], red[0]);
}

// ---------------- pass 4: finalize loss ----------------
__global__ void loss_fin(const float* __restrict__ stats, float* __restrict__ loss_out) {
    if (threadIdx.x == 0) loss_out[0] = stats[2] * (1.0f / (float)NPT);
}

extern "C" void kernel_launch(void* const* d_in, const int* in_sizes, int n_in,
                              void* d_out, int out_size, void* d_ws, size_t ws_size,
                              hipStream_t stream) {
    const float* rays  = (const float*)d_in[0];
    const float* zsamp = (const float*)d_in[1];
    // d_in[2] = invalid (all false) -- unused, matches reference semantics
    const float* W1 = (const float*)d_in[3];
    const float* b1 = (const float*)d_in[4];
    const float* W2 = (const float*)d_in[5];
    const float* b2 = (const float*)d_in[6];
    const float* W3 = (const float*)d_in[7];
    const float* b3 = (const float*)d_in[8];

    float* out   = (float*)d_out;
    float* rgb   = out;                         // 32768*3
    float* depth = out + NRAYS * 3;             // 32768
    float* wts   = out + NRAYS * 4;             // 32768*64
    float* loss  = out + NRAYS * 4 + NPT;       // 1

    float* stats = (float*)d_ws;                // 16 floats of accumulators
    float* sig   = stats + 16;                  // NPT sigmas (8 MB)

    init_stats<<<1, 64, 0, stream>>>(stats);
    nerf_fused<<<NBLK, 256, 0, stream>>>(rays, zsamp, W1, b1, W2, b2, W3, b3,
                                         rgb, depth, wts, sig, stats);
    z_pass<<<NBLK, 256, 0, stream>>>(sig, stats);
    loss_pass<<<NBLK, 256, 0, stream>>>(sig, stats);
    loss_fin<<<1, 64, 0, stream>>>(stats, loss);
}
